// Elman_11733850652808
// MI455X (gfx1250) — matrix operand order repaired and checked by reading, not compiled, not run on hardware
//
#include <hip/hip_runtime.h>

#define BB 64
#define TT 1024
#define EE 512
#define HH 512
#define OO 512

typedef __attribute__((ext_vector_type(16))) __bf16 v16bf;
typedef __attribute__((ext_vector_type(8)))  float  v8f;
typedef __attribute__((ext_vector_type(4)))  float  f32x4;
typedef __attribute__((ext_vector_type(4)))  unsigned int u32x4;
typedef __attribute__((ext_vector_type(8)))  int i32x8;
typedef __attribute__((ext_vector_type(4)))  int i32x4;

union BFrag {
    v16bf bf;
    u32x4 u[2];
    unsigned short us[16];
};

__device__ __forceinline__ unsigned short f2bf(float f) {
    unsigned u = __float_as_uint(f);
    unsigned r = (u + 0x7FFFu + ((u >> 16) & 1u)) >> 16;
    return (unsigned short)r;
}

// ---------------------------------------------------------------------------
// Pack a 512x512 f32 weight (row-major, K x N) into bf16 WMMA B-fragments.
// dst[((nt*16 + kc)*32 + lane)*16 + i] : K = kc*32 + (lane/16)*16 + i, N = nt*16 + lane%16.
// ---------------------------------------------------------------------------
__global__ void pack_w_kernel(const float* __restrict__ src,
                              unsigned short* __restrict__ dst) {
    unsigned d = blockIdx.x * blockDim.x + threadIdx.x;
    if (d >= 512u * 512u) return;
    unsigned nt   = d >> 13;
    unsigned rem  = d & 8191u;
    unsigned kc   = rem >> 9;
    unsigned rem2 = rem & 511u;
    unsigned lane = rem2 >> 4;
    unsigned i    = rem2 & 15u;
    unsigned K = kc * 32 + (lane >> 4) * 16 + i;
    unsigned n = nt * 16 + (lane & 15u);
    dst[d] = f2bf(src[(size_t)K * 512 + n]);
}

// ---------------------------------------------------------------------------
// out[row,:] = A[row,:] (f32,K=512) x Wp (bf16 packed) + bias.
// One block = 16 rows x 512 cols, 256 threads = 8 waves x 4 N-tiles.
// A tile (32 KB f32, contiguous) staged via async global->LDS DMA
// (global_load_async_to_lds_b128 / ASYNCcnt), then converted to bf16
// A-fragments in LDS.
// ---------------------------------------------------------------------------
__global__ void gemm_bias_kernel(const float* __restrict__ A,
                                 const unsigned short* __restrict__ Wp,
                                 const float* __restrict__ bias,
                                 float* __restrict__ out) {
    __shared__ float sRaw[16 * 512];              // 32 KB raw f32 A tile
    __shared__ unsigned short sA[16 * 32 * 16];   // 16 KB bf16 A fragments

    const unsigned tid = threadIdx.x;
    const unsigned long long abase =
        (unsigned long long)(A + (size_t)blockIdx.x * 16 * 512);

    // ---- async DMA: 2048 x 16B chunks, 256 threads x 8 ----
    {
#pragma unroll
        for (int i = 0; i < 8; ++i) {
            unsigned c    = tid + i * 256;
            unsigned voff = c * 16u;
            unsigned ldsa = (unsigned)(unsigned long long)
                                ((char*)sRaw + c * 16u);
            asm volatile("global_load_async_to_lds_b128 %0, %1, %2"
                         :: "v"(ldsa), "v"(voff), "s"(abase)
                         : "memory");
        }
        asm volatile("s_wait_asynccnt 0x0" ::: "memory");
    }
    __syncthreads();

    // ---- convert to bf16 A-fragments ----
    {
        const unsigned r  = tid >> 4;       // 0..15 row
        const unsigned kc = tid & 15u;      // 0..15 k-chunk
        const float* src = &sRaw[r * 512 + kc * 32];
        float v[32];
#pragma unroll
        for (int j = 0; j < 8; ++j) {
            f32x4 q = *(const f32x4*)(src + j * 4);
            v[j * 4 + 0] = q.x; v[j * 4 + 1] = q.y;
            v[j * 4 + 2] = q.z; v[j * 4 + 3] = q.w;
        }
        BFrag lo, hi;
#pragma unroll
        for (int j = 0; j < 16; ++j) lo.us[j] = f2bf(v[j]);
#pragma unroll
        for (int j = 0; j < 16; ++j) hi.us[j] = f2bf(v[16 + j]);
        *(u32x4*)&sA[(kc * 32 +      r) * 16 + 0] = lo.u[0];
        *(u32x4*)&sA[(kc * 32 +      r) * 16 + 8] = lo.u[1];
        *(u32x4*)&sA[(kc * 32 + 16 + r) * 16 + 0] = hi.u[0];
        *(u32x4*)&sA[(kc * 32 + 16 + r) * 16 + 8] = hi.u[1];
    }
    __syncthreads();

    const unsigned wave = tid >> 5;
    const unsigned lane = tid & 31u;
    const unsigned n0   = wave * 64;
    const unsigned hi   = lane >> 4;
    const unsigned nc   = lane & 15u;

    v8f acc[4];
#pragma unroll
    for (int t = 0; t < 4; ++t) {
        float bv = bias[n0 + t * 16 + nc];
#pragma unroll
        for (int i = 0; i < 8; ++i) acc[t][i] = bv;
    }

#pragma unroll
    for (int kc = 0; kc < 16; ++kc) {
        BFrag af;
        af.u[0] = *(const u32x4*)&sA[(kc * 32 + lane) * 16 + 0];
        af.u[1] = *(const u32x4*)&sA[(kc * 32 + lane) * 16 + 8];
#pragma unroll
        for (int t = 0; t < 4; ++t) {
            const unsigned nt = n0 / 16 + t;
            const unsigned short* bp = Wp + (((size_t)nt * 16 + kc) * 32 + lane) * 16;
            BFrag bf_;
            bf_.u[0] = *(const u32x4*)(bp + 0);
            bf_.u[1] = *(const u32x4*)(bp + 8);
            acc[t] = __builtin_amdgcn_wmma_f32_16x16x32_bf16(
                false, af.bf, false, bf_.bf, (short)0, acc[t], false, false);
        }
    }

#pragma unroll
    for (int t = 0; t < 4; ++t) {
#pragma unroll
        for (int r = 0; r < 8; ++r) {
            size_t row = (size_t)blockIdx.x * 16 + r + 8 * hi;
            out[row * 512 + n0 + t * 16 + nc] = acc[t][r];
        }
    }
}

// ---------------------------------------------------------------------------
// Recurrence scan. grid = 4 blocks (16 batch rows each), block = 512 (16 waves).
// Each step t: h_new = relu(xw[:,t,:] + h @ Wh) -> overwrite Hbuf[:,t,:].
// xw tiles (16x512 f32, row stride T*512*4 B) are double-buffered in LDS via
// the Tensor Data Mover: one tensor_load_to_lds per step, issued by wave 0,
// tracked with TENSORcnt, prefetched one step ahead of the serial chain.
// ---------------------------------------------------------------------------
__global__ void scan_kernel(const unsigned short* __restrict__ Whp,
                            float* __restrict__ Hbuf,
                            float* __restrict__ hfinal) {
    __shared__ unsigned short sH[16 * 32 * 16];   // 16 KB: h as bf16 fragments
    __shared__ float sXW[2][16 * 512];            // 2 x 32 KB: xw tile buffers

    const unsigned tid  = threadIdx.x;
    const unsigned wave = tid >> 5;
    const unsigned lane = tid & 31u;
    const unsigned bg   = blockIdx.x;
    const unsigned n0   = wave * 32;
    const unsigned hi   = lane >> 4;
    const unsigned nc   = lane & 15u;
    const unsigned rb   = bg * 16;

    const unsigned long long hbase = (unsigned long long)Hbuf;
    const unsigned lds0 = (unsigned)(unsigned long long)&sXW[0][0];
    const unsigned lds1 = (unsigned)(unsigned long long)&sXW[1][0];

    // ---- TDM descriptor group 1 (t-invariant) ----
    // [17:16] data_size=2 (4B); tensor_dim0=512; tensor_dim1=1<<20;
    // tile_dim0=512; tile_dim1=16; tensor_dim0_stride=T*512 elements.
    i32x8 g1;
    g1[0] = 0x00020000;   // data_size=2, workgroup_mask=0
    g1[1] = 0x02000000;   // tensor_dim0[15:0]=512 @ bits 63:48
    g1[2] = 0x00000000;   // tensor_dim0 hi=0, tensor_dim1[15:0]=0
    g1[3] = 0x02000010;   // tensor_dim1[31:16]=16 (=1<<20), tile_dim0=512
    g1[4] = 0x00000010;   // tile_dim1=16, tile_dim2=0
    g1[5] = 0x00080000;   // tensor_dim0_stride = 524288 (T*512)
    g1[6] = 0x00000000;
    g1[7] = 0x00000000;
    const i32x4 gz = {0, 0, 0, 0};

    auto tdm_issue = [&](unsigned tp, unsigned buf) {
        unsigned long long ga =
            hbase + ((unsigned long long)(rb * TT + tp) << 11);  // *512*4 bytes
        u32x4 g0;
        g0[0] = 1u;                                   // count=1, 2D, no gather
        g0[1] = buf ? lds1 : lds0;                    // LDS destination
        g0[2] = (unsigned)(ga & 0xFFFFFFFFu);         // global_addr lo
        g0[3] = (unsigned)((ga >> 32) & 0x01FFFFFFu)  // global_addr[56:32]
              | (2u << 30);                           // type=2 (image)
#if __clang_major__ >= 23
        __builtin_amdgcn_tensor_load_to_lds(g0, g1, gz, gz,
                                            (i32x8){0,0,0,0,0,0,0,0}, 0);
#else
        __builtin_amdgcn_tensor_load_to_lds(g0, g1, gz, gz, 0);
#endif
    };

    // h0 = 0
    {
        unsigned* z = (unsigned*)sH;
#pragma unroll
        for (int i = 0; i < 8; ++i) z[tid + i * 512] = 0u;
    }

    // preload tile 0 (wave 0 only; scalar branch so other waves skip issue)
    if (tid < 32) {
        tdm_issue(0u, 0u);
        __builtin_amdgcn_s_wait_tensorcnt((short)0);
    }
    __syncthreads();

    for (int t = 0; t < TT; ++t) {
        if (tid < 32) {
            // prefetch next tile into the other buffer (wraps at end; the
            // wrap iteration's data is never consumed)
            tdm_issue((unsigned)(t + 1) & (TT - 1), (unsigned)(t + 1) & 1u);
            // TDM ops retire in order: <=1 outstanding means tile t landed
            __builtin_amdgcn_s_wait_tensorcnt((short)1);
        }
        __syncthreads();   // xw(t) visible to all; sH stable from prev step

        const float* xw = &sXW[t & 1][0];
        v8f acc[2];
#pragma unroll
        for (int tt = 0; tt < 2; ++tt) {
#pragma unroll
            for (int r = 0; r < 8; ++r)
                acc[tt][r] = xw[(r + 8 * hi) * 512 + n0 + tt * 16 + nc];
        }
        // acc += h @ Wh over K=512
#pragma unroll
        for (int kc = 0; kc < 16; ++kc) {
            BFrag af;
            af.u[0] = *(const u32x4*)&sH[(kc * 32 + lane) * 16 + 0];
            af.u[1] = *(const u32x4*)&sH[(kc * 32 + lane) * 16 + 8];
#pragma unroll
            for (int tt = 0; tt < 2; ++tt) {
                const unsigned nt = wave * 2 + tt;
                const unsigned short* bp =
                    Whp + (((size_t)nt * 16 + kc) * 32 + lane) * 16;
                BFrag bf_;
                bf_.u[0] = *(const u32x4*)(bp + 0);
                bf_.u[1] = *(const u32x4*)(bp + 8);
                acc[tt] = __builtin_amdgcn_wmma_f32_16x16x32_bf16(
                    false, af.bf, false, bf_.bf, (short)0, acc[tt], false, false);
            }
        }
#pragma unroll
        for (int tt = 0; tt < 2; ++tt)
#pragma unroll
            for (int r = 0; r < 8; ++r)
                acc[tt][r] = fmaxf(acc[tt][r], 0.0f);

        __syncthreads();   // everyone done reading sH this step

#pragma unroll
        for (int tt = 0; tt < 2; ++tt) {
#pragma unroll
            for (int r = 0; r < 8; ++r) {
                float v = acc[tt][r];
                unsigned m = r + 8 * hi;
                size_t idx = (((size_t)(rb + m)) * TT + t) * 512
                           + n0 + tt * 16 + nc;
                Hbuf[idx] = v;
                unsigned kin  = tt * 16 + nc;
                unsigned la   = (kin < 16) ? m : (16 + m);
                unsigned ii   = kin & 15u;
                sH[(wave * 32 + la) * 16 + ii] = f2bf(v);
                if (t == TT - 1)
                    hfinal[(size_t)(rb + m) * 512 + n0 + tt * 16 + nc] = v;
            }
        }
        __syncthreads();   // sH fully updated before next step's reads
    }
}

extern "C" void kernel_launch(void* const* d_in, const int* in_sizes, int n_in,
                              void* d_out, int out_size, void* d_ws, size_t ws_size,
                              hipStream_t stream) {
    const float* x  = (const float*)d_in[0];   // (64,1024,512)
    const float* W1 = (const float*)d_in[1];   // (1024,512)
    const float* b1 = (const float*)d_in[2];   // (512,)
    const float* W2 = (const float*)d_in[3];   // (512,512)
    const float* b2 = (const float*)d_in[4];   // (512,)

    float* out = (float*)d_out;
    float* hfinal = out + (size_t)BB * TT * OO;

    char* w = (char*)d_ws;
    const size_t HB_BYTES = (size_t)BB * TT * HH * sizeof(float);
    float* Hbuf = (float*)w;
    unsigned short* Wxp = (unsigned short*)(w + HB_BYTES);
    unsigned short* Whp = Wxp + 512 * 512;
    unsigned short* W2p = Whp + 512 * 512;

    pack_w_kernel<<<1024, 256, 0, stream>>>(W1,             Wxp);
    pack_w_kernel<<<1024, 256, 0, stream>>>(W1 + 512 * 512, Whp);
    pack_w_kernel<<<1024, 256, 0, stream>>>(W2,             W2p);

    gemm_bias_kernel<<<(BB * TT) / 16, 256, 0, stream>>>(x, Wxp, b1, Hbuf);

    scan_kernel<<<BB / 16, 512, 0, stream>>>(Whp, Hbuf, hfinal);

    gemm_bias_kernel<<<(BB * TT) / 16, 256, 0, stream>>>(Hbuf, W2p, b2, out);
}
